// fraFace_52261162057742
// MI455X (gfx1250) — matrix operand order repaired and checked by reading, not compiled, not run on hardware
//
#include <hip/hip_runtime.h>
#include <hip/hip_bf16.h>
#include <math.h>

// Problem constants (match reference)
#define BB 1024
#define EE 512
#define CC 70722
#define M_COEF 0.4f
#define H_COEF 0.333f
#define SCALE_F 64.0f
#define EPS_F 1e-3f
#define PI_F 3.14159265358979323846f

typedef __attribute__((ext_vector_type(16))) __bf16 v16bf;
typedef __attribute__((ext_vector_type(8)))  float  v8f;

// CDNA5 async global->LDS path (guarded: falls back cleanly if absent)
#if defined(__HIP_DEVICE_COMPILE__) && \
    __has_builtin(__builtin_amdgcn_global_load_async_to_lds_b128) && \
    __has_builtin(__builtin_amdgcn_s_wait_asynccnt)
#define USE_ASYNC 1
typedef int v4i_t __attribute__((__vector_size__(16)));
typedef __attribute__((address_space(1))) v4i_t* g4_t;  // global int4*
typedef __attribute__((address_space(3))) v4i_t* l4_t;  // LDS int4*
#else
#define USE_ASYNC 0
#endif

// ---------- bf16 split helpers (round-to-nearest-even) ----------
__device__ __forceinline__ unsigned short f2bf(float f) {
  unsigned u = __builtin_bit_cast(unsigned, f);
  u += 0x7FFFu + ((u >> 16) & 1u);
  return (unsigned short)(u >> 16);
}
__device__ __forceinline__ float bf2f(unsigned short h) {
  unsigned u = ((unsigned)h) << 16;
  return __builtin_bit_cast(float, u);
}

// ---------- 1) column inverse L2 norms of kernel (E x C) ----------
__global__ void colnorm_k(const float* __restrict__ W, float* __restrict__ inv_norm) {
  int c = blockIdx.x * 256 + threadIdx.x;
  if (c >= CC) return;
  float s = 0.f;
  for (int e = 0; e < EE; ++e) {
    float v = W[(size_t)e * CC + c];
    s = fmaf(v, v, s);
  }
  inv_norm[c] = rsqrtf(fmaxf(s, 1e-30f));
}

// ---------- 2) normalize + split kernel into bf16 hi/lo ----------
__global__ void wconv_k(const float* __restrict__ W, const float* __restrict__ inv_norm,
                        unsigned short* __restrict__ khi, unsigned short* __restrict__ klo) {
  int c = blockIdx.x * 256 + threadIdx.x;
  int e = blockIdx.y;
  if (c >= CC) return;
  size_t idx = (size_t)e * CC + c;
  float v = W[idx] * inv_norm[c];
  unsigned short h = f2bf(v);
  khi[idx] = h;
  klo[idx] = f2bf(v - bf2f(h));
}

// ---------- 3) split embeddings into bf16 hi/lo ----------
__global__ void aconv_k(const float* __restrict__ A,
                        unsigned short* __restrict__ ahi, unsigned short* __restrict__ alo) {
  int col = blockIdx.x * 256 + threadIdx.x;
  int row = blockIdx.y;
  int idx = row * EE + col;
  float v = A[idx];
  unsigned short h = f2bf(v);
  ahi[idx] = h;
  alo[idx] = f2bf(v - bf2f(h));
}

// ---------- 4) batch stats over norms -> per-row margins ----------
__global__ void stats_k(const float* __restrict__ norms,
                        float* __restrict__ gang, float* __restrict__ gadd) {
  __shared__ float ssum[256];
  __shared__ float ssq[256];
  int t = threadIdx.x;
  float x[4];
  float s = 0.f, q = 0.f;
#pragma unroll
  for (int i = 0; i < 4; ++i) {
    float v = norms[t * 4 + i];
    v = fminf(fmaxf(v, 1e-3f), 100.f);
    x[i] = v;
    s += v;
    q = fmaf(v, v, q);
  }
  ssum[t] = s; ssq[t] = q;
  __syncthreads();
  for (int off = 128; off > 0; off >>= 1) {
    if (t < off) { ssum[t] += ssum[t + off]; ssq[t] += ssq[t + off]; }
    __syncthreads();
  }
  float mean = ssum[0] / (float)BB;
  float var  = (ssq[0] - (float)BB * mean * mean) / (float)(BB - 1);
  float stdv = sqrtf(fmaxf(var, 0.f));
#pragma unroll
  for (int i = 0; i < 4; ++i) {
    float ms = (x[i] - mean) / (stdv + EPS_F) * H_COEF;
    ms = fminf(fmaxf(ms, -1.f), 1.f);
    gang[t * 4 + i] = -M_COEF * ms;
    gadd[t * 4 + i] = M_COEF + M_COEF * ms;
  }
}

// ---------- 5) split-bf16 WMMA GEMM + fused AdaFace epilogue ----------
#define TM 128
#define TN 128
#define TK 32
#define LDA 40  // ushort stride of A tile rows (M x K, K-contiguous, padded)
#define LDB 40  // ushort stride of Bs tile rows (N x K transposed, K-contiguous, padded)

__global__ __launch_bounds__(256, 1) void gemm_k(
    const unsigned short* __restrict__ Ahi, const unsigned short* __restrict__ Alo,
    const unsigned short* __restrict__ Khi, const unsigned short* __restrict__ Klo,
    const int* __restrict__ label, const float* __restrict__ gang,
    const float* __restrict__ gadd, float* __restrict__ out) {
  __shared__ unsigned short As[2][TM * LDA];  // [hi/lo] row-major M x K
  __shared__ unsigned short Bs[2][TN * LDB];  // [hi/lo] transposed: N x K
  __shared__ int   lab_s[TM];
  __shared__ float ga_s[TM];
  __shared__ float gd_s[TM];

  const int tid  = threadIdx.x;
  const int lane = tid & 31;
  const int wave = tid >> 5;
  const int m_base = blockIdx.y * TM;
  const int n_base = blockIdx.x * TN;

  if (tid < TM) {
    int r = m_base + tid;
    lab_s[tid] = label[r];
    ga_s[tid]  = gang[r];
    gd_s[tid]  = gadd[r];
  }

  const int wm = (wave & 1) * 64;  // wave M offset inside tile
  const int wn = (wave >> 1) * 32; // wave N offset inside tile

  v8f acc[4][2];
#pragma unroll
  for (int mi = 0; mi < 4; ++mi)
#pragma unroll
    for (int ni = 0; ni < 2; ++ni)
      acc[mi][ni] = (v8f){0.f, 0.f, 0.f, 0.f, 0.f, 0.f, 0.f, 0.f};

  const int a_row  = tid >> 1;        // 0..127
  const int a_half = (tid & 1) * 16;  // 0 / 16 (ushorts)
  const int k4 = (tid >> 5) * 4;      // B staging: 4 K rows per wave
  const int n4 = (tid & 31) * 4;      // B staging: 4 N cols per lane

  for (int kk = 0; kk < EE; kk += TK) {
    // ---- stage A tile (hi & lo): async DMA to LDS when available ----
    {
      const size_t ga = (size_t)(m_base + a_row) * EE + kk + a_half;
#if USE_ASYNC
      // GLOBAL_LOAD_ASYNC_TO_LDS_B128 x4 (ASYNCcnt-tracked)
      __builtin_amdgcn_global_load_async_to_lds_b128(
          (g4_t)(Ahi + ga), (l4_t)&As[0][a_row * LDA + a_half], 0, 0);
      __builtin_amdgcn_global_load_async_to_lds_b128(
          (g4_t)(Ahi + ga + 8), (l4_t)&As[0][a_row * LDA + a_half + 8], 0, 0);
      __builtin_amdgcn_global_load_async_to_lds_b128(
          (g4_t)(Alo + ga), (l4_t)&As[1][a_row * LDA + a_half], 0, 0);
      __builtin_amdgcn_global_load_async_to_lds_b128(
          (g4_t)(Alo + ga + 8), (l4_t)&As[1][a_row * LDA + a_half + 8], 0, 0);
#else
      const uint4* ph = (const uint4*)(Ahi + ga);
      const uint4* pl = (const uint4*)(Alo + ga);
      uint4* dh = (uint4*)&As[0][a_row * LDA + a_half];
      uint4* dl = (uint4*)&As[1][a_row * LDA + a_half];
      dh[0] = ph[0]; dh[1] = ph[1];
      dl[0] = pl[0]; dl[1] = pl[1];
      if (kk + TK < EE) {
        __builtin_prefetch(Ahi + ga + TK, 0, 3);  // global_prefetch_b8
        __builtin_prefetch(Alo + ga + TK, 0, 3);
      }
#endif
    }

    // ---- stage B tile transposed: each thread moves a 4(K) x 4(N) block ----
    {
      const int gcol = n_base + n4;
      const int rem = CC - gcol;
      unsigned rh[4][2], rl[4][2];  // [k row j][2 dwords = 4 ushorts along n]
      if (rem >= 4) {
#pragma unroll
        for (int j = 0; j < 4; ++j) {
          const size_t gb = (size_t)(kk + k4 + j) * CC + gcol;
          const unsigned* ph = (const unsigned*)(Khi + gb);
          const unsigned* pl = (const unsigned*)(Klo + gb);
          rh[j][0] = ph[0]; rh[j][1] = ph[1];
          rl[j][0] = pl[0]; rl[j][1] = pl[1];
        }
        if (kk + TK < EE) {
          __builtin_prefetch(Khi + (size_t)(kk + TK + k4) * CC + gcol, 0, 1);
          __builtin_prefetch(Klo + (size_t)(kk + TK + k4) * CC + gcol, 0, 1);
        }
      } else {
#pragma unroll
        for (int j = 0; j < 4; ++j) {
          const size_t gb = (size_t)(kk + k4 + j) * CC + gcol;
          unsigned short th[4], tl[4];
#pragma unroll
          for (int c = 0; c < 4; ++c) {
            th[c] = (c < rem) ? Khi[gb + c] : (unsigned short)0;
            tl[c] = (c < rem) ? Klo[gb + c] : (unsigned short)0;
          }
          rh[j][0] = th[0] | ((unsigned)th[1] << 16);
          rh[j][1] = th[2] | ((unsigned)th[3] << 16);
          rl[j][0] = tl[0] | ((unsigned)tl[1] << 16);
          rl[j][1] = tl[2] | ((unsigned)tl[3] << 16);
        }
      }
      // transpose 4x4: pack K-pairs per column, store b64 per column
#pragma unroll
      for (int c = 0; c < 4; ++c) {
        const int half = c >> 1;
        const int sh = (c & 1) * 16;
        unsigned w0 = ((rh[0][half] >> sh) & 0xffffu) | (((rh[1][half] >> sh) & 0xffffu) << 16);
        unsigned w1 = ((rh[2][half] >> sh) & 0xffffu) | (((rh[3][half] >> sh) & 0xffffu) << 16);
        *(uint2*)&Bs[0][(n4 + c) * LDB + k4] = make_uint2(w0, w1);
        unsigned v0 = ((rl[0][half] >> sh) & 0xffffu) | (((rl[1][half] >> sh) & 0xffffu) << 16);
        unsigned v1 = ((rl[2][half] >> sh) & 0xffffu) | (((rl[3][half] >> sh) & 0xffffu) << 16);
        *(uint2*)&Bs[1][(n4 + c) * LDB + k4] = make_uint2(v0, v1);
      }
    }

#if USE_ASYNC
    __builtin_amdgcn_s_wait_asynccnt(0);
#endif
    __syncthreads();

    // ---- assemble fragments per ISA 16-bit VGPR layouts (all b128 LDS loads) ----
    const int ar = lane & 15;
    const int kh = (lane >> 4) << 3;  // A: K chunks {kh..kh+7, kh+16..kh+23}
    const int kb = (lane >> 4) << 3;  // B: K chunks {kb..kb+7, kb+16..kb+23}

    v16bf afh[4], afl[4];
#pragma unroll
    for (int mi = 0; mi < 4; ++mi) {
      union { uint4 u[2]; v16bf v; } t;
      const unsigned short* pr = &As[0][(wm + mi * 16 + ar) * LDA];
      t.u[0] = *(const uint4*)(pr + kh);
      t.u[1] = *(const uint4*)(pr + kh + 16);
      afh[mi] = t.v;
      const unsigned short* pr2 = &As[1][(wm + mi * 16 + ar) * LDA];
      t.u[0] = *(const uint4*)(pr2 + kh);
      t.u[1] = *(const uint4*)(pr2 + kh + 16);
      afl[mi] = t.v;
    }
    v16bf bfh[2], bfl[2];
#pragma unroll
    for (int ni = 0; ni < 2; ++ni) {
      union { uint4 u[2]; v16bf v; } t;
      const unsigned short* pn0 = &Bs[0][(wn + ni * 16 + ar) * LDB];
      t.u[0] = *(const uint4*)(pn0 + kb);
      t.u[1] = *(const uint4*)(pn0 + kb + 16);
      bfh[ni] = t.v;
      const unsigned short* pn1 = &Bs[1][(wn + ni * 16 + ar) * LDB];
      t.u[0] = *(const uint4*)(pn1 + kb);
      t.u[1] = *(const uint4*)(pn1 + kb + 16);
      bfl[ni] = t.v;
    }

    // ---- split-bf16 product: hi*hi + hi*lo + lo*hi ----
#pragma unroll
    for (int mi = 0; mi < 4; ++mi)
#pragma unroll
      for (int ni = 0; ni < 2; ++ni) {
        acc[mi][ni] = __builtin_amdgcn_wmma_f32_16x16x32_bf16(
            false, afh[mi], false, bfh[ni], (short)0, acc[mi][ni], false, false);
        acc[mi][ni] = __builtin_amdgcn_wmma_f32_16x16x32_bf16(
            false, afh[mi], false, bfl[ni], (short)0, acc[mi][ni], false, false);
        acc[mi][ni] = __builtin_amdgcn_wmma_f32_16x16x32_bf16(
            false, afl[mi], false, bfh[ni], (short)0, acc[mi][ni], false, false);
      }
    __syncthreads();
  }

  // ---- fused AdaFace epilogue ----
  const float cosClip = 1.0f - EPS_F;
#pragma unroll
  for (int ni = 0; ni < 2; ++ni) {
    const int n = n_base + wn + ni * 16 + (lane & 15);
    if (n >= CC) continue;
#pragma unroll
    for (int mi = 0; mi < 4; ++mi) {
      const int rb = wm + mi * 16 + ((lane >> 4) << 3);
      v8f cc = acc[mi][ni];
#pragma unroll
      for (int i = 0; i < 8; ++i) {
        const int r = rb + i;  // local row in [0,128)
        float v = cc[i];
        v = fminf(fmaxf(v, -cosClip), cosClip);
        float th = acosf(v);
        const bool hit = (lab_s[r] == n);
        if (hit) th += ga_s[r];
        th = fminf(fmaxf(th, EPS_F), PI_F - EPS_F);
        float cm = __cosf(th);
        if (hit) cm -= gd_s[r];
        out[(size_t)(m_base + r) * CC + n] = cm * SCALE_F;
      }
    }
  }
}

extern "C" void kernel_launch(void* const* d_in, const int* in_sizes, int n_in,
                              void* d_out, int out_size, void* d_ws, size_t ws_size,
                              hipStream_t stream) {
  (void)in_sizes; (void)n_in; (void)out_size; (void)ws_size;
  const float* emb   = (const float*)d_in[0];
  const float* norms = (const float*)d_in[1];
  const int*   label = (const int*)d_in[2];
  const float* W     = (const float*)d_in[3];
  float* out = (float*)d_out;

  char* ws = (char*)d_ws;
  size_t off = 0;
  auto alloc = [&](size_t bytes) -> char* {
    char* p = ws + off;
    off = (off + bytes + 255) & ~(size_t)255;
    return p;
  };
  float* inv_norm     = (float*)alloc((size_t)CC * 4);
  float* gang         = (float*)alloc((size_t)BB * 4);
  float* gadd         = (float*)alloc((size_t)BB * 4);
  unsigned short* ahi = (unsigned short*)alloc((size_t)BB * EE * 2);
  unsigned short* alo = (unsigned short*)alloc((size_t)BB * EE * 2);
  unsigned short* khi = (unsigned short*)alloc((size_t)EE * CC * 2);
  unsigned short* klo = (unsigned short*)alloc((size_t)EE * CC * 2);

  colnorm_k<<<dim3((CC + 255) / 256), dim3(256), 0, stream>>>(W, inv_norm);
  wconv_k<<<dim3((CC + 255) / 256, EE), dim3(256), 0, stream>>>(W, inv_norm, khi, klo);
  aconv_k<<<dim3(EE / 256, BB), dim3(256), 0, stream>>>(emb, ahi, alo);
  stats_k<<<dim3(1), dim3(256), 0, stream>>>(norms, gang, gadd);
  gemm_k<<<dim3((CC + TN - 1) / TN, BB / TM), dim3(256), 0, stream>>>(
      ahi, alo, khi, klo, label, gang, gadd, out);
}
